// MultiHeadAttention_75591424409842
// MI455X (gfx1250) — matrix operand order repaired and checked by reading, not compiled, not run on hardware
//
#include <hip/hip_runtime.h>
#include <hip/hip_bf16.h>
#include <math.h>

typedef __attribute__((ext_vector_type(16))) _Float16 v16h;
typedef __attribute__((ext_vector_type(8)))  _Float16 v8h;
typedef __attribute__((ext_vector_type(8)))  float    v8f;

#define D_MODEL 1024
#define SEQ     2048
#define NB      2
#define NH      16
#define HD      64
#define NTOK    (NB * SEQ)          // 4096 token rows
#define NELT_X  (NTOK * D_MODEL)    // 4,194,304
#define NELT_W  (D_MODEL * D_MODEL) // 1,048,576
#define KC      128                 // k-chunk staged per async round
#define NCHUNK  (D_MODEL / KC)      // 8

static __device__ __forceinline__ v8h ld8(const _Float16* p) {
  return *(const v8h*)p;
}
static __device__ __forceinline__ v16h ld16(const _Float16* p) {
  return *(const v16h*)p;
}
static __device__ __forceinline__ v16h mk16(v8h lo, v8h hi) {
  return __builtin_shufflevector(lo, hi, 0, 1, 2, 3, 4, 5, 6, 7,
                                 8, 9, 10, 11, 12, 13, 14, 15);
}
static __device__ __forceinline__ v8f zero8() {
  v8f z;
  for (int i = 0; i < 8; ++i) z[i] = 0.0f;
  return z;
}
static __device__ __forceinline__ v8f wmma_f16(v16h a, v16h b, v8f c) {
  return __builtin_amdgcn_wmma_f32_16x16x32_f16(false, a, false, b,
                                                (short)0, c, false, false);
}

// ---------------------------------------------------------------------------
// Elementwise f32 -> f16 convert (for x).
// ---------------------------------------------------------------------------
__global__ void cvt_f16_kernel(const float* __restrict__ src,
                               _Float16* __restrict__ dst, int n) {
  int i = blockIdx.x * blockDim.x + threadIdx.x;
  if (i < n) dst[i] = (_Float16)src[i];
}

// ---------------------------------------------------------------------------
// Weight transpose + convert: Wt[n][k] = (f16) W[k][n]   (1024x1024)
// ---------------------------------------------------------------------------
__global__ void transpose_cvt_kernel(const float* __restrict__ W,
                                     _Float16* __restrict__ Wt) {
  int i = blockIdx.x * blockDim.x + threadIdx.x; // 0 .. 1M-1
  int k = i >> 10;
  int n = i & 1023;
  Wt[(size_t)n * D_MODEL + k] = (_Float16)W[i];
}

// ---------------------------------------------------------------------------
// Async-stage one KC-wide chunk of the 64-column B panel into LDS.
// Block-wide: 128 threads x 8 x b128 = 16KB (64 n-rows x 128 k, f16).
// Uses GLOBAL_LOAD_ASYNC_TO_LDS_B128 (tracked by ASYNCcnt): 8 async
// instructions per wave per chunk.
// ---------------------------------------------------------------------------
static __device__ __forceinline__ void stage_async(
    const _Float16* __restrict__ Bt, int c0, int k0, _Float16* bufPtr) {
  const int tid = threadIdx.x; // 0..127
#pragma unroll
  for (int itr = 0; itr < 8; ++itr) {
    int vid = itr * 128 + tid;        // 0..1023 vectors of 8 f16
    int n   = vid >> 4;               // 0..63
    int kv  = (vid & 15) * 8;         // 0..120
    unsigned long long g =
        (unsigned long long)(size_t)(Bt + (size_t)(c0 + n) * D_MODEL + k0 + kv);
    unsigned lds = (unsigned)(size_t)(bufPtr + n * KC + kv);
    asm volatile("global_load_async_to_lds_b128 %0, %1, off"
                 :: "v"(lds), "v"(g) : "memory");
  }
}

// ---------------------------------------------------------------------------
// Per-wave MMA over one staged KC chunk: acc[4] += A[16xKC] * Bsh[KCx64].
// A from global (row-major), B fragments from LDS (contiguous 32B per lane).
// ---------------------------------------------------------------------------
static __device__ __forceinline__ void mma_chunk(
    const _Float16* __restrict__ arow, const _Float16* bsh,
    int hs, int r, v8f acc[4]) {
#pragma unroll
  for (int ks = 0; ks < KC; ks += 32) {
    v16h af = mk16(ld8(arow + ks + hs * 16), ld8(arow + ks + hs * 16 + 8));
#pragma unroll
    for (int t = 0; t < 4; ++t) {
      v16h bf = ld16(bsh + (t * 16 + r) * KC + ks + hs * 16);
      acc[t] = wmma_f16(af, bf, acc[t]);
    }
  }
}

// ---------------------------------------------------------------------------
// Double-buffered async GEMM body shared by qkv / oproj kernels.
// Each 4-wave block: 4 row tiles (one per wave) x 64 shared columns.
// ---------------------------------------------------------------------------
static __device__ __forceinline__ void gemm_body(
    const _Float16* __restrict__ A, const _Float16* __restrict__ Bt,
    _Float16* Bsh /* [2][64*KC] */, int i0, int c0, v8f acc[4]) {
  const int lane = threadIdx.x & 31;
  const int r = lane & 15, hs = lane >> 4;
  const _Float16* arow = A + (size_t)(i0 + r) * D_MODEL;

  stage_async(Bt, c0, 0, Bsh);
  stage_async(Bt, c0, KC, Bsh + 64 * KC);

#pragma unroll
  for (int c = 0; c < NCHUNK; ++c) {
    if (c == NCHUNK - 1)
      asm volatile("s_wait_asynccnt 0" ::: "memory");
    else
      asm volatile("s_wait_asynccnt 8" ::: "memory");
    __syncthreads();                       // chunk c visible to all waves
    __builtin_prefetch(arow + (c + 1) * KC, 0, 0);
    mma_chunk(arow + c * KC, Bsh + (c & 1) * 64 * KC, hs, r, acc);
    __syncthreads();                       // done reading buffer (c&1)
    if (c + 2 < NCHUNK)
      stage_async(Bt, c0, (c + 2) * KC, Bsh + (c & 1) * 64 * KC);
  }
}

// ---------------------------------------------------------------------------
// Fused QKV projection + RoPE epilogue.
//   block -> (matrix m, 64-col group, group of 4 row tiles; 1 tile per wave)
//   Q: scaled by 1/sqrt(HD), RoPE on first 32 dims/head, f16 [b][h][s][64]
//   K: RoPE, f16 [b][h][s][64]
//   V: TRANSPOSED f16 [b][h][64][s]  (one v8h store per lane)
// ---------------------------------------------------------------------------
__global__ void __launch_bounds__(128) qkv_kernel(
    const _Float16* __restrict__ xb,
    const _Float16* __restrict__ wqt, const _Float16* __restrict__ wkt,
    const _Float16* __restrict__ wvt,
    _Float16* __restrict__ Qb, _Float16* __restrict__ Kb,
    _Float16* __restrict__ Vt) {
  __shared__ __align__(32) _Float16 Bsh[2][64 * KC]; // 32KB double buffer
  const int blk = blockIdx.x;          // 0 .. 3071
  const int w   = threadIdx.x >> 5;
  const int rtg = blk & 63;            // 64 groups of 4 row tiles
  const int cg  = (blk >> 6) & 15;     // 16 col groups
  const int m   = blk >> 10;           // 0=q 1=k 2=v
  const int rt  = rtg * 4 + w;
  const int i0 = rt * 16, c0 = cg * 64;
  const _Float16* Bt = (m == 0) ? wqt : ((m == 1) ? wkt : wvt);

  v8f acc[4];
#pragma unroll
  for (int t = 0; t < 4; ++t) acc[t] = zero8();
  gemm_body(xb, Bt, &Bsh[0][0], i0, c0, acc);

  const int lane = threadIdx.x & 31;
  const int r = lane & 15, hs = lane >> 4;
  const float qs = (m == 0) ? 0.125f : 1.0f; // 1/sqrt(64)

#pragma unroll
  for (int t = 0; t < 4; ++t) {
    int c  = c0 + t * 16 + r;
    int h  = c >> 6;
    int kd = c & 63;
    if (m < 2) {
      bool inRope = (kd < 32);
      float invf = __powf(10000.0f, -((float)(kd & ~1)) * (1.0f / 32.0f));
      _Float16* dst = (m == 0) ? Qb : Kb;
#pragma unroll
      for (int e = 0; e < 8; ++e) {
        int tok = i0 + e + 8 * hs;
        int b = tok >> 11, s = tok & (SEQ - 1);
        float val = acc[t][e] * qs;
        float par = __shfl_xor(val, 1, 32); // partner feature c^1
        float o = val;
        if (inRope) {
          float ang = (float)s * invf;
          float cs = __cosf(ang), sn = __sinf(ang);
          o = (kd & 1) ? (val * cs + par * sn) : (val * cs - par * sn);
        }
        size_t idx = (((size_t)b * NH + h) * SEQ + s) * HD + kd;
        dst[idx] = (_Float16)o;
      }
    } else {
      v8h vv;
#pragma unroll
      for (int e = 0; e < 8; ++e) vv[e] = (_Float16)acc[t][e];
      int tok0 = i0 + 8 * hs;
      int b = tok0 >> 11, s0 = tok0 & (SEQ - 1);
      size_t idx = (((size_t)b * NH + h) * HD + kd) * SEQ + s0;
      *(v8h*)(Vt + idx) = vv;
    }
  }
}

// ---------------------------------------------------------------------------
// Flash attention: one wave per (b,h, 16-row query block).
// Key tiles of 32, online softmax in f32, P transposed C->A layout through
// a 1KB per-wave LDS buffer, PV accumulated with 4 WMMAs per step.
// ---------------------------------------------------------------------------
__global__ void __launch_bounds__(128) attn_kernel(
    const _Float16* __restrict__ Qb, const _Float16* __restrict__ Kb,
    const _Float16* __restrict__ Vt, _Float16* __restrict__ Y) {
  __shared__ __align__(32) _Float16 pbuf[4][16 * 32];
  const int w   = threadIdx.x >> 5;
  const int wid = blockIdx.x * 4 + w;   // 0 .. 4095
  const int bh  = wid >> 7;             // 0 .. 31
  const int it  = wid & 127;
  const int i0  = it * 16;
  const int b   = bh >> 4, h = bh & 15;
  const _Float16* Q = Qb + (size_t)bh * SEQ * HD;
  const _Float16* K = Kb + (size_t)bh * SEQ * HD;
  const _Float16* V = Vt + (size_t)bh * HD * SEQ;
  _Float16* plds = pbuf[w];
  const int lane = threadIdx.x & 31;
  const int r = lane & 15, hs = lane >> 4;

  v16h qf[2];
  {
    const _Float16* qrow = Q + (size_t)(i0 + r) * HD;
#pragma unroll
    for (int kc = 0; kc < 2; ++kc)
      qf[kc] = mk16(ld8(qrow + kc * 32 + hs * 16),
                    ld8(qrow + kc * 32 + hs * 16 + 8));
  }

  v8f acc[4];
#pragma unroll
  for (int t = 0; t < 4; ++t) acc[t] = zero8();
  float mrow[8], lrow[8];
#pragma unroll
  for (int e = 0; e < 8; ++e) { mrow[e] = -3.0e38f; lrow[e] = 0.0f; }

  for (int j0 = 0; j0 <= i0; j0 += 32) {
    v8f sc[2];
#pragma unroll
    for (int jt = 0; jt < 2; ++jt) {
      v8f c = zero8();
      const _Float16* krow = K + (size_t)(j0 + jt * 16 + r) * HD;
      __builtin_prefetch(krow + 64 * HD, 0, 0);
#pragma unroll
      for (int kc = 0; kc < 2; ++kc) {
        v16h kf = ld16(krow + kc * 32 + hs * 16);
        c = wmma_f16(qf[kc], kf, c);
      }
      sc[jt] = c;
    }

    float p0[8], p1[8];
#pragma unroll
    for (int e = 0; e < 8; ++e) {
      int i  = i0 + e + 8 * hs;
      int jA = j0 + r;
      int jB = j0 + 16 + r;
      float a  = (jA <= i) ? sc[0][e] : -1.0e30f;
      float bb = (jB <= i) ? sc[1][e] : -1.0e30f;
      float rmax = fmaxf(a, bb);
#pragma unroll
      for (int ml = 1; ml < 16; ml <<= 1)
        rmax = fmaxf(rmax, __shfl_xor(rmax, ml, 16));
      float newm = fmaxf(mrow[e], rmax);
      float corr = __expf(mrow[e] - newm);
      mrow[e] = newm;
      float pa = __expf(a - newm);
      float pb = __expf(bb - newm);
      float rsum = pa + pb;
#pragma unroll
      for (int ml = 1; ml < 16; ml <<= 1)
        rsum += __shfl_xor(rsum, ml, 16);
      lrow[e] = lrow[e] * corr + rsum;
#pragma unroll
      for (int t = 0; t < 4; ++t) acc[t][e] *= corr;
      p0[e] = pa;
      p1[e] = pb;
    }

#pragma unroll
    for (int e = 0; e < 8; ++e) {
      int M = e + 8 * hs;
      plds[M * 32 + r]      = (_Float16)p0[e];
      plds[M * 32 + 16 + r] = (_Float16)p1[e];
    }
    asm volatile("s_wait_dscnt 0" ::: "memory");
    v16h pf = mk16(ld8(plds + r * 32 + hs * 16),
                   ld8(plds + r * 32 + hs * 16 + 8));

#pragma unroll
    for (int t = 0; t < 4; ++t) {
      const _Float16* vrow = V + (size_t)(t * 16 + r) * SEQ + j0;
      v16h vf = ld16(vrow + hs * 16);
      acc[t] = wmma_f16(pf, vf, acc[t]);
    }
  }

#pragma unroll
  for (int t = 0; t < 4; ++t) {
#pragma unroll
    for (int e = 0; e < 8; ++e) {
      int tok = b * SEQ + i0 + e + 8 * hs;
      int col = h * HD + t * 16 + r;
      Y[(size_t)tok * D_MODEL + col] = (_Float16)(acc[t][e] / lrow[e]);
    }
  }
}

// ---------------------------------------------------------------------------
// Output projection: d_out = Y @ Wo   (f16 in, f32 out), async-staged B.
// ---------------------------------------------------------------------------
__global__ void __launch_bounds__(128) oproj_kernel(
    const _Float16* __restrict__ Y, const _Float16* __restrict__ wot,
    float* __restrict__ out) {
  __shared__ __align__(32) _Float16 Bsh[2][64 * KC];
  const int blk = blockIdx.x;      // 0 .. 1023
  const int w   = threadIdx.x >> 5;
  const int rtg = blk & 63;
  const int cg  = blk >> 6;        // 0..15
  const int rt  = rtg * 4 + w;
  const int i0 = rt * 16, c0 = cg * 64;

  v8f acc[4];
#pragma unroll
  for (int t = 0; t < 4; ++t) acc[t] = zero8();
  gemm_body(Y, wot, &Bsh[0][0], i0, c0, acc);

  const int lane = threadIdx.x & 31;
  const int r = lane & 15, hs = lane >> 4;
#pragma unroll
  for (int t = 0; t < 4; ++t) {
#pragma unroll
    for (int e = 0; e < 8; ++e) {
      out[(size_t)(i0 + e + 8 * hs) * D_MODEL + c0 + t * 16 + r] = acc[t][e];
    }
  }
}

// ---------------------------------------------------------------------------
// Workspace layout (bytes):
//   [ 0,  8M) xb f16 | [8M,16M) Wq^T/Wk^T/Wv^T/Wo^T f16 (2M each)
//   [16M,24M) Q | [24M,32M) K | [32M,40M) V^T | [40M,48M) Y   (all f16)
// ---------------------------------------------------------------------------
extern "C" void kernel_launch(void* const* d_in, const int* in_sizes, int n_in,
                              void* d_out, int out_size, void* d_ws,
                              size_t ws_size, hipStream_t stream) {
  (void)in_sizes; (void)n_in; (void)out_size; (void)ws_size;
  const float* x  = (const float*)d_in[0];
  // d_in[1] is the causal mask (bool) -- causality is hardcoded, unused.
  const float* Wq = (const float*)d_in[2];
  const float* Wk = (const float*)d_in[3];
  const float* Wv = (const float*)d_in[4];
  const float* Wo = (const float*)d_in[5];

  char* ws = (char*)d_ws;
  const size_t MB = (size_t)1 << 20;
  _Float16* xb  = (_Float16*)(ws + 0);
  _Float16* wqt = (_Float16*)(ws + 8 * MB);
  _Float16* wkt = (_Float16*)(ws + 10 * MB);
  _Float16* wvt = (_Float16*)(ws + 12 * MB);
  _Float16* wot = (_Float16*)(ws + 14 * MB);
  _Float16* Qb  = (_Float16*)(ws + 16 * MB);
  _Float16* Kb  = (_Float16*)(ws + 24 * MB);
  _Float16* Vt  = (_Float16*)(ws + 32 * MB);
  _Float16* Yb  = (_Float16*)(ws + 40 * MB);

  cvt_f16_kernel<<<(NELT_X + 255) / 256, 256, 0, stream>>>(x, xb, NELT_X);
  transpose_cvt_kernel<<<NELT_W / 256, 256, 0, stream>>>(Wq, wqt);
  transpose_cvt_kernel<<<NELT_W / 256, 256, 0, stream>>>(Wk, wkt);
  transpose_cvt_kernel<<<NELT_W / 256, 256, 0, stream>>>(Wv, wvt);
  transpose_cvt_kernel<<<NELT_W / 256, 256, 0, stream>>>(Wo, wot);

  // 3 matrices x 16 col groups x 64 row-tile groups = 3072 blocks (4 waves)
  qkv_kernel<<<3072, 128, 0, stream>>>(xb, wqt, wkt, wvt, Qb, Kb, Vt);
  // 32 (b,h) x 128 row tiles = 4096 waves
  attn_kernel<<<1024, 128, 0, stream>>>(Qb, Kb, Vt, Yb);
  // 16 col groups x 64 row-tile groups = 1024 blocks
  oproj_kernel<<<1024, 128, 0, stream>>>(Yb, wot, (float*)d_out);
}
